// NODE_noise_64544768525255
// MI455X (gfx1250) — matrix operand order repaired and checked by reading, not compiled, not run on hardware
//
#include <hip/hip_runtime.h>
#include <hip/hip_bf16.h>

// ---------------------------------------------------------------------------
// Problem constants (from reference)
// ---------------------------------------------------------------------------
#define B_   256
#define NMOD 3
#define R_   32
#define M_   512
#define D_   98            // NMOD*R + 2
#define KU   96            // NMOD*R
#define TSZ  76033         // B + B*96 + M*D + 2M + 1
#define OUTW 76034         // 1 + TSZ
#define UOFF 257           // 1 + B
#define SOFF 24833         // UOFF + B*96
#define WOFF 75009         // SOFF + M*D
#define ZPER 24833         // per-row zero count: cols [1,24833) + last col

typedef __attribute__((ext_vector_type(16))) _Float16 v16h;
typedef __attribute__((ext_vector_type(8)))  float    v8f;

static __device__ __forceinline__ float2 block_reduce2(float a, float b, float2* lds) {
    int tid = threadIdx.x;
    lds[tid] = make_float2(a, b);
    __syncthreads();
    for (int off = 256; off >= 1; off >>= 1) {
        if (tid < off) {
            lds[tid].x += lds[tid + off].x;
            lds[tid].y += lds[tid + off].y;
        }
        __syncthreads();
    }
    float2 r = lds[0];
    __syncthreads();
    return r;
}

// ---------------------------------------------------------------------------
// 0) zero ONLY the regions that stay zero: per row b, the contiguous span
//    cols [1, 24833) (beta block + U region; diagonals rewritten later by
//    k_rk4 / k_wmma_ju) plus the final log_tau column. 26 MB instead of 78.
// ---------------------------------------------------------------------------
__global__ void k_zero_sparse(float* __restrict__ out) {
    int idx = blockIdx.x * blockDim.x + threadIdx.x;
    if (idx >= B_ * ZPER) return;
    int b = idx / ZPER;
    int r = idx - b * ZPER;
    size_t col = (r == ZPER - 1) ? (size_t)(OUTW - 1) : (size_t)(1 + r);
    __builtin_nontemporal_store(0.f, &out[(size_t)b * OUTW + col]);
}

// ---------------------------------------------------------------------------
// 1) gather Uvec rows + beta = sum_r prod_i U_i   (256 blocks x 96 threads)
// ---------------------------------------------------------------------------
__global__ void k_gather(const int* __restrict__ bin,
                         const float* __restrict__ U0,
                         const float* __restrict__ U1,
                         const float* __restrict__ U2,
                         float* __restrict__ Uv, _Float16* __restrict__ Uvh,
                         float* __restrict__ beta) {
    __shared__ float lds[KU];
    int b = blockIdx.x, t = threadIdx.x;
    int i = t >> 5, r = t & 31;
    const float* Up = (i == 0) ? U0 : ((i == 1) ? U1 : U2);
    int idx = bin[b * NMOD + i];
    float v = Up[idx * R_ + r];
    Uv[b * KU + t]  = v;
    Uvh[b * KU + t] = (_Float16)v;
    lds[t] = v;
    __syncthreads();
    if (t == 0) {
        float s = 0.f;
        for (int rr = 0; rr < R_; ++rr) s += lds[rr] * lds[R_ + rr] * lds[2 * R_ + rr];
        beta[b] = s;
    }
}

// ---------------------------------------------------------------------------
// 2) Hbase[b,m] = sum_j Uv[b,j]*S[m,2+j]  -- 256x512 GEMM, K=96, WMMA f16
//    one wave per 16x16 tile; 16 btiles x 32 ntiles = 512 blocks of 32 thr
// ---------------------------------------------------------------------------
__global__ void k_wmma_hbase(const _Float16* __restrict__ Uvh,
                             const float* __restrict__ S,
                             float* __restrict__ Hbase) {
    int tile = blockIdx.x;
    int bt = tile >> 5, nt = tile & 31;
    int lane = threadIdx.x;
    bool hi = lane >= 16;
    int rowA = bt * 16 + (lane & 15);
    int colB = nt * 16 + (lane & 15);
    v8f acc = {};
#pragma unroll
    for (int k0 = 0; k0 < KU; k0 += 32) {
        v16h a, bf;
#pragma unroll
        for (int t = 0; t < 16; ++t) {
            // A 16x32, contiguous k = 16h + t (matches B)
            int ka = k0 + t + (hi ? 16 : 0);
            a[t] = Uvh[rowA * KU + ka];
            // B 32x16, contiguous k = 16h + t (matches A); B[k][n] = S[n, 2+k]
            int kb = k0 + t + (hi ? 16 : 0);
            bf[t] = (_Float16)S[colB * D_ + 2 + kb];
        }
        acc = __builtin_amdgcn_wmma_f32_16x16x32_f16(false, a, false, bf,
                                                     (short)0, acc, false, false);
    }
#pragma unroll
    for (int v = 0; v < 8; ++v) {
        int row = bt * 16 + v + (hi ? 8 : 0);
        Hbase[row * M_ + colB] = acc[v];
    }
}

// ---------------------------------------------------------------------------
// 3) RK4 over the scalar x-ODE + closed-form sensitivity coefficients.
//    One block (512 threads, one per m) per batch. Since dg = a*g + J with
//    per-batch scalar a, g_final = A*g0 + s1*J1+s2*J2+s3*J3+s4*J4.
// ---------------------------------------------------------------------------
__global__ void k_rk4(const float* __restrict__ btn,
                      const float* __restrict__ beta,
                      const float* __restrict__ S,
                      const float* __restrict__ w,
                      const float* __restrict__ Hbase,
                      float* __restrict__ Cc, _Float16* __restrict__ Cch,
                      float* __restrict__ v0, float* __restrict__ v1,
                      float* __restrict__ out) {
    __shared__ float2 lds[M_];
    int b = blockIdx.x, m = threadIdx.x;
    const float inv = 0.04419417382415922f;           // 1/sqrt(512)
    float scaleT = btn[b];
    float x0 = beta[b];
    float sm0 = S[m * D_ + 0], sm1 = S[m * D_ + 1];
    float hb = Hbase[b * M_ + m];
    float wc = w[m], wsn = w[M_ + m];

    const float ts[4] = {0.f, 0.5f, 0.5f, 1.f};
    float cs[4], sn[4], cc[4], fs[4], as[4], xs[4];
    xs[0] = x0;
    for (int i = 0; i < 4; ++i) {
        float h = sm0 * scaleT * ts[i] + sm1 * xs[i] + hb;
        float sv, cv;
        __sincosf(h, &sv, &cv);
        cs[i] = cv; sn[i] = sv;
        cc[i] = cv * wsn - sv * wc;                   // d(cos*wc+sin*ws)/dh
        float2 r = block_reduce2(cv * wc + sv * wsn, cc[i] * sm1, lds);
        fs[i] = scaleT * inv * r.x;                   // dx at stage i
        as[i] = scaleT * inv * r.y;                   // df/dx at stage i
        if (i == 0)      xs[1] = x0 + 0.5f * fs[0];
        else if (i == 1) xs[2] = x0 + 0.5f * fs[1];
        else if (i == 2) xs[3] = x0 + fs[2];
    }
    float xf = x0 + (fs[0] + 2.f * fs[1] + 2.f * fs[2] + fs[3]) * (1.f / 6.f);
    float a1 = as[0], a2 = as[1], a3 = as[2], a4 = as[3];
    // RK4 linear-propagation coefficients (h=1)
    float s4 = 1.f / 6.f;
    float s3 = (2.f + a4) / 6.f;
    float s2 = (2.f + a3 + 0.5f * a3 * a4) / 6.f;
    float s1 = (1.f + a2 + 0.5f * a2 * a3 + 0.25f * a2 * a3 * a4) / 6.f;
    float k2A = a2 * (1.f + 0.5f * a1);
    float k3A = a3 * (1.f + 0.5f * k2A);
    float k4A = a4 * (1.f + k3A);
    float Af = 1.f + (a1 + 2.f * k2A + 2.f * k3A + k4A) / 6.f;

    float scale = scaleT * inv;
    float Ccv = scale * (s1 * cc[0] + s2 * cc[1] + s3 * cc[2] + s4 * cc[3]);
    float v0v = scaleT * scale * (0.5f * s2 * cc[1] + 0.5f * s3 * cc[2] + s4 * cc[3]);
    float v1v = scale * (s1 * cc[0] * xs[0] + s2 * cc[1] * xs[1] +
                         s3 * cc[2] * xs[2] + s4 * cc[3] * xs[3]);
    float wcs = scale * (s1 * cs[0] + s2 * cs[1] + s3 * cs[2] + s4 * cs[3]);
    float wsv = scale * (s1 * sn[0] + s2 * sn[1] + s3 * sn[2] + s4 * sn[3]);

    int bm = b * M_ + m;
    Cc[bm] = Ccv; Cch[bm] = (_Float16)Ccv; v0[bm] = v0v; v1[bm] = v1v;
    size_t ro = (size_t)b * OUTW;
    out[ro + WOFF + m]      = wcs;                    // grad wrt w[m]
    out[ro + WOFF + M_ + m] = wsv;                    // grad wrt w[M+m]
    if (m == 0) {
        out[ro]         = xf;                         // integrated x
        out[ro + 1 + b] = Af;                         // diag of beta-block
    }
}

// ---------------------------------------------------------------------------
// 4) JU[b,j] = sum_m Cc[b,m]*S[m,2+j] -- 256x96 GEMM, K=512, WMMA f16.
//    Scatter into the block-diagonal U-grad region of the output.
// ---------------------------------------------------------------------------
__global__ void k_wmma_ju(const _Float16* __restrict__ Cch,
                          const float* __restrict__ S,
                          float* __restrict__ out) {
    int tile = blockIdx.x;                             // 16 btiles x 6 ntiles
    int bt = tile / 6, nt = tile % 6;
    int lane = threadIdx.x;
    bool hi = lane >= 16;
    int rowA = bt * 16 + (lane & 15);
    int colB = nt * 16 + (lane & 15);
    v8f acc = {};
#pragma unroll 4
    for (int k0 = 0; k0 < M_; k0 += 32) {
        v16h a, bf;
#pragma unroll
        for (int t = 0; t < 16; ++t) {
            int ka = k0 + t + (hi ? 16 : 0);
            a[t] = Cch[rowA * M_ + ka];
            int kb = k0 + t + (hi ? 16 : 0);
            bf[t] = (_Float16)S[kb * D_ + 2 + colB];   // B[k][n] = S[k,2+n]
        }
        acc = __builtin_amdgcn_wmma_f32_16x16x32_f16(false, a, false, bf,
                                                     (short)0, acc, false, false);
    }
#pragma unroll
    for (int v = 0; v < 8; ++v) {
        int row = bt * 16 + v + (hi ? 8 : 0);
        out[(size_t)row * OUTW + UOFF + row * KU + colB] = acc[v];
    }
}

// ---------------------------------------------------------------------------
// 5) S-gradient block: out[b, SOFF + m*98 + k] = rank-4 combo. The region per
//    batch is one contiguous 50176-float span (r = m*98+k), so flat element
//    indexing gives fully-coalesced, 100%-utilized non-temporal stores.
// ---------------------------------------------------------------------------
__global__ void k_sgrad(const float* __restrict__ Cc,
                        const float* __restrict__ v0,
                        const float* __restrict__ v1,
                        const float* __restrict__ Uv,
                        float* __restrict__ out) {
    int idx = blockIdx.x * blockDim.x + threadIdx.x;
    if (idx >= B_ * M_ * D_) return;
    int b = idx / (M_ * D_);
    int r = idx - b * (M_ * D_);
    int m = r / D_;
    int k = r - m * D_;
    int bm = b * M_ + m;
    float val;
    if (k == 0)      val = v0[bm];
    else if (k == 1) val = v1[bm];
    else             val = Cc[bm] * Uv[b * KU + (k - 2)];
    __builtin_nontemporal_store(val, &out[(size_t)b * OUTW + SOFF + r]);
}

// ---------------------------------------------------------------------------
extern "C" void kernel_launch(void* const* d_in, const int* in_sizes, int n_in,
                              void* d_out, int out_size, void* d_ws, size_t ws_size,
                              hipStream_t stream) {
    const int*   bin = (const int*)  d_in[0];  // (B,3)
    const float* btn = (const float*)d_in[1];  // (B,)
    const float* U0  = (const float*)d_in[2];
    const float* U1  = (const float*)d_in[3];
    const float* U2  = (const float*)d_in[4];
    const float* S   = (const float*)d_in[5];  // (M,D)
    const float* w   = (const float*)d_in[6];  // (2M,)
    float* out = (float*)d_out;

    char* ws = (char*)d_ws;
    float*     Uv    = (float*)(ws);                       // B*96
    float*     Hbase = (float*)(ws + 98304);               // B*M
    float*     Cc    = (float*)(ws + 622592);              // B*M
    float*     v0    = (float*)(ws + 1146880);             // B*M
    float*     v1    = (float*)(ws + 1671168);             // B*M
    float*     beta  = (float*)(ws + 2195456);             // B
    _Float16*  Uvh   = (_Float16*)(ws + 2196480);          // B*96
    _Float16*  Cch   = (_Float16*)(ws + 2245632);          // B*M

    int nz = B_ * ZPER;
    k_zero_sparse<<<(nz + 255) / 256, 256, 0, stream>>>(out);
    k_gather<<<B_, KU, 0, stream>>>(bin, U0, U1, U2, Uv, Uvh, beta);
    k_wmma_hbase<<<512, 32, 0, stream>>>(Uvh, S, Hbase);
    k_rk4<<<B_, M_, 0, stream>>>(btn, beta, S, w, Hbase, Cc, Cch, v0, v1, out);
    k_wmma_ju<<<96, 32, 0, stream>>>(Cch, S, out);
    int ns = B_ * M_ * D_;
    k_sgrad<<<(ns + 255) / 256, 256, 0, stream>>>(Cc, v0, v1, Uv, out);
}
